// Model_83141976916196
// MI455X (gfx1250) — matrix operand order repaired and checked by reading, not compiled, not run on hardware
//
#include <hip/hip_runtime.h>
#include <hip/hip_bf16.h>

typedef __attribute__((ext_vector_type(16))) __bf16 v16bf;
typedef __attribute__((ext_vector_type(8)))  float  v8f;
typedef unsigned short u16;
typedef unsigned int   u32;
typedef int i32x4 __attribute__((__vector_size__(4 * sizeof(int))));

#define B_  4
#define T_  2048
#define C_  768
#define H_  8
#define HS_ 96
#define M_  (B_ * T_)      // 8192
#define N3_ (3 * C_)       // 2304

#ifndef __has_builtin
#define __has_builtin(x) 0
#endif

#if __has_builtin(__builtin_amdgcn_global_load_async_to_lds_b128)
#define USE_ASYNC_LDS 1
#else
#define USE_ASYNC_LDS 0
#endif

#define AS1 __attribute__((address_space(1)))
#define AS3 __attribute__((address_space(3)))

// ---------------------------------------------------------------- helpers

__device__ __forceinline__ u16 f2bf(float f) {
    union { float f; u32 u; } x; x.f = f;
    u32 r = (x.u + 0x7FFFu + ((x.u >> 16) & 1u)) >> 16;
    return (u16)r;
}

// Build a 16x(K=32) bf16 fragment from two 16-byte chunks (8 bf16 each).
__device__ __forceinline__ v16bf load_frag2(const u16* p0, const u16* p1) {
    union { uint4 q[2]; v16bf v; } u;
    u.q[0] = *(const uint4*)p0;
    u.q[1] = *(const uint4*)p1;
    return u.v;
}

__device__ __forceinline__ v8f wmma_bf16(v16bf a, v16bf b, v8f c) {
    return __builtin_amdgcn_wmma_f32_16x16x32_bf16(
        /*neg_a=*/false, a, /*neg_b=*/false, b,
        /*c_mod=*/(short)0, c, /*reuse_a=*/false, /*reuse_b=*/false);
}

__device__ __forceinline__ void wait_async_le12() {
#if __has_builtin(__builtin_amdgcn_s_wait_asynccnt)
    __builtin_amdgcn_s_wait_asynccnt(12);
#else
    asm volatile("s_wait_asynccnt 12" ::: "memory");
#endif
    asm volatile("" ::: "memory");
}
__device__ __forceinline__ void wait_async_0() {
#if __has_builtin(__builtin_amdgcn_s_wait_asynccnt)
    __builtin_amdgcn_s_wait_asynccnt(0);
#else
    asm volatile("s_wait_asynccnt 0" ::: "memory");
#endif
    asm volatile("" ::: "memory");
}

// reduce across the 16 lanes sharing a D-matrix row group (wave32 halves)
__device__ __forceinline__ float red_max16(float v) {
    v = fmaxf(v, __shfl_xor(v, 1, 32));
    v = fmaxf(v, __shfl_xor(v, 2, 32));
    v = fmaxf(v, __shfl_xor(v, 4, 32));
    v = fmaxf(v, __shfl_xor(v, 8, 32));
    return v;
}
__device__ __forceinline__ float red_sum16(float v) {
    v += __shfl_xor(v, 1, 32);
    v += __shfl_xor(v, 2, 32);
    v += __shfl_xor(v, 4, 32);
    v += __shfl_xor(v, 8, 32);
    return v;
}

// ---------------------------------------------------------------- precision prep

__global__ void k_f32_to_bf16(const float* __restrict__ src, u16* __restrict__ dst, int n) {
    int i = blockIdx.x * 256 + threadIdx.x;
    if (i < n) dst[i] = f2bf(src[i]);
}

// src: fp32 [K][N] row-major -> dst: bf16 [N][K]
__global__ void k_transpose_bf16(const float* __restrict__ src, u16* __restrict__ dst,
                                 int K, int N) {
    int i = blockIdx.x * 256 + threadIdx.x;
    if (i < K * N) {
        int n = i % N, k = i / N;
        dst[n * K + k] = f2bf(src[i]);
    }
}

// ---------------------------------------------------------------- QKV GEMM
// xb:[M,C] bf16, waT:[3C,C] bf16 (transposed weights). Scatter into
// q,k: [B,H,T,hs] bf16 ; v stored transposed: [B,H,hs,T] bf16.
__global__ __launch_bounds__(256) void k_qkv_gemm(
    const u16* __restrict__ xb, const u16* __restrict__ waT,
    const float* __restrict__ b_attn,
    u16* __restrict__ qbuf, u16* __restrict__ kbuf, u16* __restrict__ vT)
{
    const int lane = threadIdx.x & 31, wave = threadIdx.x >> 5;
    const int lg = lane >> 4, ln = lane & 15;
    const int m0 = blockIdx.x * 64  + (wave >> 2) * 32;   // 2 wave-rows
    const int n0 = blockIdx.y * 256 + (wave & 3) * 64;    // 4 wave-cols

    v8f acc[2][4];
#pragma unroll
    for (int i = 0; i < 2; ++i)
#pragma unroll
        for (int j = 0; j < 4; ++j) acc[i][j] = {};

    for (int k0 = 0; k0 < C_; k0 += 32) {
        v16bf a[2], b[4];
#pragma unroll
        for (int i = 0; i < 2; ++i) {
            const u16* base = xb + (m0 + 16 * i + ln) * C_ + k0 + 8 * lg;
            a[i] = load_frag2(base, base + 16);
        }
#pragma unroll
        for (int j = 0; j < 4; ++j) {
            const u16* base = waT + (n0 + 16 * j + ln) * C_ + k0 + 8 * lg;
            b[j] = load_frag2(base, base + 16);
        }
#pragma unroll
        for (int i = 0; i < 2; ++i)
#pragma unroll
            for (int j = 0; j < 4; ++j)
                acc[i][j] = wmma_bf16(a[i], b[j], acc[i][j]);
    }

#pragma unroll
    for (int i = 0; i < 2; ++i)
#pragma unroll
        for (int j = 0; j < 4; ++j) {
            const int n = n0 + 16 * j + ln;
            const int which = n / C_;
            const int c = n % C_;
            const int h = c / HS_, d = c % HS_;
            const float bias = b_attn[n];
#pragma unroll
            for (int r = 0; r < 8; ++r) {
                const int m = m0 + 16 * i + r + 8 * lg;
                const int bb = m >> 11, t = m & (T_ - 1);
                const u16 bv = f2bf(acc[i][j][r] + bias);
                if (which == 0)
                    qbuf[((bb * H_ + h) * T_ + t) * HS_ + d] = bv;
                else if (which == 1)
                    kbuf[((bb * H_ + h) * T_ + t) * HS_ + d] = bv;
                else
                    vT[((bb * H_ + h) * HS_ + d) * T_ + t] = bv;
            }
        }
}

// ---------------------------------------------------------------- flash attention
// One block = 4 waves, each wave owns 16 query rows. grid = B*H*(T/64).
// K-tiles are double-buffered in LDS via ASYNC loads (prefetch next tile
// while computing current); V consumed directly from global (contiguous b128).
__global__ __launch_bounds__(128) void k_attn(
    const u16* __restrict__ qbuf, const u16* __restrict__ kbuf,
    const u16* __restrict__ vT, u16* __restrict__ yb)
{
    __shared__ u16 sP[4][16 * 32];        // per-wave P tile (C-layout -> A-layout)
#if USE_ASYNC_LDS
    __shared__ u16 sK[4][2][32 * HS_];    // per-wave double-buffered K tile (6KB ea)
#endif

    const int lane = threadIdx.x & 31, wave = threadIdx.x >> 5;
    const int lg = lane >> 4, ln = lane & 15;
    const int bh = blockIdx.x >> 5;          // b*H + h
    const int qt = blockIdx.x & 31;
    const int q0 = qt * 64 + wave * 16;

    const u16* Qp = qbuf + (size_t)bh * T_ * HS_;
    const u16* Kp = kbuf + (size_t)bh * T_ * HS_;
    const u16* Vp = vT   + (size_t)bh * HS_ * T_;

    const int steps = (q0 + 16 + 31) >> 5;     // causal: keys [0, q0+15]

#if USE_ASYNC_LDS
    // async-copy one contiguous 32x96 bf16 K tile (6144B = 384 x b128 chunks)
    auto issue_ktile = [&](int kb, int buf) {
        const u16* g = Kp + (size_t)kb * HS_;
        u16* l = &sK[wave][buf][0];
#pragma unroll
        for (int it = 0; it < 12; ++it) {
            const int idx = lane + 32 * it;           // chunk id, 8 u16 per chunk
            __builtin_amdgcn_global_load_async_to_lds_b128(
                (AS1 i32x4*)(g + idx * 8), (AS3 i32x4*)(l + idx * 8), 0, 0);
        }
    };
    issue_ktile(0, 0);   // prefetch first tile before anything else
#endif

    // resident Q fragments: 16 rows x hs=96 (3 chunks of K=32)
    v16bf aQ[3];
#pragma unroll
    for (int c = 0; c < 3; ++c) {
        const u16* base = Qp + (q0 + ln) * HS_ + 32 * c + 8 * lg;
        aQ[c] = load_frag2(base, base + 16);
    }

    v8f accO[6];
#pragma unroll
    for (int c = 0; c < 6; ++c) accO[c] = {};
    float mrun[8], lrun[8];
#pragma unroll
    for (int r = 0; r < 8; ++r) { mrun[r] = -1e30f; lrun[r] = 0.0f; }

    const float scale = 0.10206207261596577f;  // 1/sqrt(96)

    for (int s = 0; s < steps; ++s) {
        const int kb = s * 32;

#if USE_ASYNC_LDS
        // prefetch next tile into the other buffer, then wait for current tile
        if (s + 1 < steps) { issue_ktile(kb + 32, (s + 1) & 1); wait_async_le12(); }
        else               { wait_async_0(); }
        const u16* Kt = &sK[wave][s & 1][0];
#endif

        // S = Q @ K^T : two 16x16 tiles covering 32 keys
        v8f s0 = {}, s1 = {};
#pragma unroll
        for (int c = 0; c < 3; ++c) {
#if USE_ASYNC_LDS
            const u16* kb0 = Kt + ln * HS_        + 32 * c + 8 * lg;
            const u16* kb1 = Kt + (16 + ln) * HS_ + 32 * c + 8 * lg;
#else
            const u16* kb0 = Kp + (kb + ln) * HS_      + 32 * c + 8 * lg;
            const u16* kb1 = Kp + (kb + 16 + ln) * HS_ + 32 * c + 8 * lg;
#endif
            v16bf b0 = load_frag2(kb0, kb0 + 16);
            v16bf b1 = load_frag2(kb1, kb1 + 16);
            s0 = wmma_bf16(aQ[c], b0, s0);
            s1 = wmma_bf16(aQ[c], b1, s1);
        }

        const int key0 = kb + ln, key1 = key0 + 16;
#pragma unroll
        for (int r = 0; r < 8; ++r) {
            const int qidx = q0 + r + 8 * lg;
            float v0 = s0[r] * scale; if (key0 > qidx) v0 = -1e30f;
            float v1 = s1[r] * scale; if (key1 > qidx) v1 = -1e30f;

            const float mx   = red_max16(fmaxf(v0, v1));
            const float mnew = fmaxf(mrun[r], mx);
            const float alpha = __expf(mrun[r] - mnew);
            const float p0 = __expf(v0 - mnew);
            const float p1 = __expf(v1 - mnew);
            const float rs = red_sum16(p0 + p1);
            lrun[r] = lrun[r] * alpha + rs;
            mrun[r] = mnew;
#pragma unroll
            for (int c2 = 0; c2 < 6; ++c2) accO[c2][r] *= alpha;

            const int row = r + 8 * lg;                 // C-layout row
            sP[wave][row * 32 + ln]      = f2bf(p0);
            sP[wave][row * 32 + 16 + ln] = f2bf(p1);
        }
        asm volatile("s_wait_dscnt 0x0" ::: "memory");

        // P back as A-fragment (16 q-rows x 32 keys)
        v16bf aP;
        {
            const u16* base = &sP[wave][ln * 32 + 8 * lg];
            aP = load_frag2(base, base + 16);
        }

        // O += P @ V  (V transposed in memory: contiguous along keys)
#pragma unroll
        for (int c2 = 0; c2 < 6; ++c2) {
            const u16* vb = Vp + (16 * c2 + ln) * T_ + kb + 8 * lg;
            v16bf bV = load_frag2(vb, vb + 16);
            accO[c2] = wmma_bf16(aP, bV, accO[c2]);
        }
    }

    // epilogue: divide by (sum + eps), write y[b, t, h*hs + d] (bf16)
    const int b = bh >> 3, h = bh & 7;
#pragma unroll
    for (int r = 0; r < 8; ++r) {
        const float inv = 1.0f / (lrun[r] + 1e-6f);
        const int t = q0 + r + 8 * lg;
#pragma unroll
        for (int c2 = 0; c2 < 6; ++c2) {
            const int col = h * HS_ + 16 * c2 + ln;
            yb[((size_t)(b * T_ + t)) * C_ + col] = f2bf(accO[c2][r] * inv);
        }
    }
}

// ---------------------------------------------------------------- output proj
// out(f32)[M,C] = yb(bf16)[M,C] @ wpT(bf16, transposed) + b_proj
__global__ __launch_bounds__(256) void k_proj_gemm(
    const u16* __restrict__ yb, const u16* __restrict__ wpT,
    const float* __restrict__ b_proj, float* __restrict__ out)
{
    const int lane = threadIdx.x & 31, wave = threadIdx.x >> 5;
    const int lg = lane >> 4, ln = lane & 15;
    const int m0 = blockIdx.x * 64  + (wave >> 2) * 32;
    const int n0 = blockIdx.y * 256 + (wave & 3) * 64;

    v8f acc[2][4];
#pragma unroll
    for (int i = 0; i < 2; ++i)
#pragma unroll
        for (int j = 0; j < 4; ++j) acc[i][j] = {};

    for (int k0 = 0; k0 < C_; k0 += 32) {
        v16bf a[2], b[4];
#pragma unroll
        for (int i = 0; i < 2; ++i) {
            const u16* base = yb + (m0 + 16 * i + ln) * C_ + k0 + 8 * lg;
            a[i] = load_frag2(base, base + 16);
        }
#pragma unroll
        for (int j = 0; j < 4; ++j) {
            const u16* base = wpT + (n0 + 16 * j + ln) * C_ + k0 + 8 * lg;
            b[j] = load_frag2(base, base + 16);
        }
#pragma unroll
        for (int i = 0; i < 2; ++i)
#pragma unroll
            for (int j = 0; j < 4; ++j)
                acc[i][j] = wmma_bf16(a[i], b[j], acc[i][j]);
    }

#pragma unroll
    for (int i = 0; i < 2; ++i)
#pragma unroll
        for (int j = 0; j < 4; ++j) {
            const int n = n0 + 16 * j + ln;
            const float bias = b_proj[n];
#pragma unroll
            for (int r = 0; r < 8; ++r) {
                const int m = m0 + 16 * i + r + 8 * lg;
                out[(size_t)m * C_ + n] = acc[i][j][r] + bias;
            }
        }
}

// ---------------------------------------------------------------- launch

extern "C" void kernel_launch(void* const* d_in, const int* in_sizes, int n_in,
                              void* d_out, int out_size, void* d_ws, size_t ws_size,
                              hipStream_t stream) {
    const float* x      = (const float*)d_in[0];
    const float* w_attn = (const float*)d_in[1];
    const float* b_attn = (const float*)d_in[2];
    const float* w_proj = (const float*)d_in[3];
    const float* b_proj = (const float*)d_in[4];
    float* out = (float*)d_out;

    char* ws = (char*)d_ws;
    size_t off = 0;
    auto alloc = [&](size_t bytes) -> void* {
        void* p = ws + off;
        off += (bytes + 255) & ~(size_t)255;
        return p;
    };
    u16* xb   = (u16*)alloc((size_t)M_ * C_ * 2);      // x in bf16
    u16* waT  = (u16*)alloc((size_t)N3_ * C_ * 2);     // w_attn^T bf16
    u16* wpT  = (u16*)alloc((size_t)C_ * C_ * 2);      // w_proj^T bf16
    u16* qbuf = (u16*)alloc((size_t)B_ * H_ * T_ * HS_ * 2);
    u16* kbuf = (u16*)alloc((size_t)B_ * H_ * T_ * HS_ * 2);
    u16* vT   = (u16*)alloc((size_t)B_ * H_ * HS_ * T_ * 2);
    u16* yb   = (u16*)alloc((size_t)M_ * C_ * 2);

    // 1. precision prep
    {
        int n = M_ * C_;
        k_f32_to_bf16<<<(n + 255) / 256, 256, 0, stream>>>(x, xb, n);
    }
    {
        int n = C_ * N3_;
        k_transpose_bf16<<<(n + 255) / 256, 256, 0, stream>>>(w_attn, waT, C_, N3_);
    }
    {
        int n = C_ * C_;
        k_transpose_bf16<<<(n + 255) / 256, 256, 0, stream>>>(w_proj, wpT, C_, C_);
    }

    // 2. QKV projection + head reorder (V transposed)
    k_qkv_gemm<<<dim3(M_ / 64, N3_ / 256), 256, 0, stream>>>(
        xb, waT, b_attn, qbuf, kbuf, vT);

    // 3. causal flash attention (async K-tile pipeline)
    k_attn<<<B_ * H_ * (T_ / 64), 128, 0, stream>>>(qbuf, kbuf, vT, yb);

    // 4. output projection
    k_proj_gemm<<<dim3(M_ / 64, C_ / 256), 256, 0, stream>>>(yb, wpT, b_proj, out);
}
